// Local_Module_19138374271385
// MI455X (gfx1250) — matrix operand order repaired and checked by reading, not compiled, not run on hardware
//
#include <hip/hip_runtime.h>
#include <math.h>

// ---------------------------------------------------------------------------
// Types for WMMA / vector loads
// ---------------------------------------------------------------------------
typedef __attribute__((ext_vector_type(16))) _Float16 v16h;
typedef __attribute__((ext_vector_type(8)))  float    v8f;
typedef __attribute__((ext_vector_type(4)))  float    v4f;

#define NNODES 20000
#define NEDGES 480000
#define HDIM   128
#define INDIM  128

// ws layout (bytes)
#define WP_FRAGS   384                         // w1:64  w2:32  w3:32  wi:128  wo:128
#define WP_BYTES   (WP_FRAGS * 512 * 2)        // 393216
#define DENOM_OFF  WP_BYTES                    // 20000 * 4
#define ATT_OFF    (DENOM_OFF + 80000)         // 473216, 16B aligned
#define DH_OFF     (ATT_OFF + NEDGES * 4)      // 2393216, 16B aligned
// total ~12.63 MB

// fragment base indices inside wp
#define FR_W1 0
#define FR_W2 64
#define FR_W3 96
#define FR_WI 128
#define FR_WO 256

__device__ __forceinline__ v8f wmma_f16(v16h a, v16h b, v8f c) {
  return __builtin_amdgcn_wmma_f32_16x16x32_f16(false, a, false, b,
                                                (short)0, c, false, false);
}

// A-matrix f16 16x32 fragment from a row-major LDS tile (ISA 7.12.2 layout):
//  lane<16 : M=lane,    VGPR j<4 -> K = kbase+2j,     j>=4 -> K = kbase+16+2(j-4)
//  lane>=16: M=lane-16, VGPR j<4 -> K = kbase+8+2j,   j>=4 -> K = kbase+24+2(j-4)
// Each half (j<4, j>=4) is 8 contiguous halfs -> lowers to ds_load_b128 x2.
__device__ __forceinline__ v16h load_a_frag(const _Float16* lds, int stride,
                                            int kbase, int lane) {
  const int hl = lane >> 4, m = lane & 15;
  v16h a;
#pragma unroll
  for (int j = 0; j < 8; ++j) {
    int k = kbase + ((j < 4) ? (hl * 8 + 2 * j) : (16 + hl * 8 + 2 * (j - 4)));
    a[2 * j]     = lds[m * stride + k];
    a[2 * j + 1] = lds[m * stride + k + 1];
  }
  return a;
}

// B fragment pre-packed per-lane contiguous: 512 halfs per fragment,
// lane gets 16 halfs at wp + frag*512 + lane*16  (two b128 loads)
__device__ __forceinline__ v16h load_b_frag(const _Float16* wp, int frag,
                                            int lane) {
  const uint4* p =
      reinterpret_cast<const uint4*>(wp + (size_t)frag * 512) + lane * 2;
  union { uint4 u[2]; v16h v; } bb;
  bb.u[0] = p[0];
  bb.u[1] = p[1];
  return bb.v;
}

__device__ __forceinline__ float leaky(float x) {
  return x >= 0.f ? x : 0.01f * x;
}

// ---------------------------------------------------------------------------
// K0: convert weights to f16, pre-swizzled into WMMA B-fragment order.
// B 32x16 f16 layout: lane<16 -> N=lane,    K = kbase+{0..7,16..23} (VGPR j: 2j,2j+1)
//                     lane>=16-> N=lane-16, K = kbase+{8..15,24..31}
// frag id = matrix_base + kc * (N/16) + nt
// ---------------------------------------------------------------------------
__global__ void pack_weights(const float* w1, const float* w2, const float* w3,
                             const float* wi, const float* wo, _Float16* wp) {
  const int wave = (int)((blockIdx.x * blockDim.x + threadIdx.x) >> 5);
  const int lane = (int)(threadIdx.x & 31);
  if (wave >= WP_FRAGS) return;
  const float* src; int Nn, local;
  if      (wave < FR_W2) { src = w1; Nn = 128; local = wave - FR_W1; }
  else if (wave < FR_W3) { src = w2; Nn = 128; local = wave - FR_W2; }
  else if (wave < FR_WI) { src = w3; Nn = 128; local = wave - FR_W3; }
  else if (wave < FR_WO) { src = wi; Nn = 512; local = wave - FR_WI; }
  else                   { src = wo; Nn = 128; local = wave - FR_WO; }
  const int ntN = Nn / 16;
  const int kc = local / ntN, nt = local % ntN;
  const int hl = lane >> 4;
  const int n  = nt * 16 + (lane & 15);
  _Float16* dst = wp + (size_t)wave * 512 + lane * 16;
#pragma unroll
  for (int j = 0; j < 8; ++j) {
    int k = kc * 32 + ((j < 4) ? (hl * 8 + 2 * j) : (16 + hl * 8 + 2 * (j - 4)));
    dst[2 * j]     = (_Float16)src[(size_t)k * Nn + n];
    dst[2 * j + 1] = (_Float16)src[(size_t)(k + 1) * Nn + n];
  }
}

// ---------------------------------------------------------------------------
// K1: one wave per edge. att = exp(sigmoid(leaky(message . A)));
// atomicAdd into denom[dst].  h_E loads are non-temporal (pure streaming).
// ---------------------------------------------------------------------------
__global__ void att_kernel(const float* __restrict__ h_V,
                           const float* __restrict__ h_E,
                           const float* __restrict__ Av,
                           const int* __restrict__ dst,
                           float* __restrict__ att,
                           float* __restrict__ denom) {
  const int e    = (int)((blockIdx.x * blockDim.x + threadIdx.x) >> 5);
  const int lane = (int)(threadIdx.x & 31);
  const int d = dst[e];
  float s = 0.f;
#pragma unroll
  for (int j = 0; j < 4; ++j) {
    int c = lane + 32 * j;
    s += h_V[(size_t)d * HDIM + c] * Av[c];
    s += __builtin_nontemporal_load(h_E + (size_t)e * INDIM + c) * Av[128 + c];
  }
#pragma unroll
  for (int off = 16; off >= 1; off >>= 1) s += __shfl_xor(s, off);
  if (lane == 0) {
    float lr = leaky(s);
    float a  = expf(1.f / (1.f + expf(-lr)));
    att[e] = a;
    atomicAdd(denom + d, a);
  }
}

// ---------------------------------------------------------------------------
// K2: one wave per 64-edge tile (4 M sub-tiles share every B fragment ->
// 4x less weight traffic from L2).  message -> GEMM1 -> GEMM2 -> GEMM3,
// scale by att/denom/SCALE, scatter-add into dh.
// One wave per block; LDS = 49.6 KB.
// ---------------------------------------------------------------------------
__global__ void __launch_bounds__(32)
edge_kernel(const float* __restrict__ h_V, const float* __restrict__ h_E,
            const int* __restrict__ dst, const float* __restrict__ b1,
            const float* __restrict__ b2, const float* __restrict__ b3,
            const _Float16* __restrict__ wp, const float* __restrict__ att,
            const float* __restrict__ denom, float* __restrict__ dh) {
  const int lane = (int)(threadIdx.x & 31);
  const int tile = (int)blockIdx.x;              // 7500 tiles of 64 edges
  const int hl = lane >> 4, ln = lane & 15;

  __shared__ _Float16 s_msg[64 * 256];           // 32 KB
  __shared__ _Float16 s_h1[64 * 128];            // 16 KB
  __shared__ float    s_scale[64];
  __shared__ int      s_dst[64];
  _Float16* msg = s_msg;
  _Float16* h1  = s_h1;

  // gather message tile -> LDS f16 (64 x 256); h_E is streamed non-temporally
  for (int m = 0; m < 64; ++m) {
    int e = tile * 64 + m;
    int d = dst[e];
    float4 v = *reinterpret_cast<const float4*>(h_V + (size_t)d * HDIM + lane * 4);
    v4f   ee = __builtin_nontemporal_load(
        reinterpret_cast<const v4f*>(h_E + (size_t)e * INDIM + lane * 4));
    _Float16* row = msg + m * 256;
    row[lane * 4 + 0] = (_Float16)v.x;  row[lane * 4 + 1] = (_Float16)v.y;
    row[lane * 4 + 2] = (_Float16)v.z;  row[lane * 4 + 3] = (_Float16)v.w;
    row[128 + lane * 4 + 0] = (_Float16)ee[0]; row[128 + lane * 4 + 1] = (_Float16)ee[1];
    row[128 + lane * 4 + 2] = (_Float16)ee[2]; row[128 + lane * 4 + 3] = (_Float16)ee[3];
  }
#pragma unroll
  for (int t = 0; t < 2; ++t) {
    int m = lane + 32 * t;
    int e = tile * 64 + m;
    int d = dst[e];
    s_dst[m]   = d;
    s_scale[m] = att[e] / (denom[d] * 30.0f);
  }
  __syncthreads();

  // GEMM1: [64x256] @ w1 -> h1 [64x128], leaky relu
  for (int nt = 0; nt < 8; ++nt) {
    v8f acc[4];
#pragma unroll
    for (int s = 0; s < 4; ++s) acc[s] = v8f{0.f,0.f,0.f,0.f,0.f,0.f,0.f,0.f};
#pragma unroll
    for (int kc = 0; kc < 8; ++kc) {
      v16h b = load_b_frag(wp, FR_W1 + kc * 8 + nt, lane);
#pragma unroll
      for (int s = 0; s < 4; ++s) {
        v16h a = load_a_frag(msg + s * 16 * 256, 256, kc * 32, lane);
        acc[s] = wmma_f16(a, b, acc[s]);
      }
    }
    float bias = b1[nt * 16 + ln];
#pragma unroll
    for (int s = 0; s < 4; ++s)
#pragma unroll
      for (int r = 0; r < 8; ++r)
        h1[(s * 16 + r + hl * 8) * 128 + nt * 16 + ln] =
            (_Float16)leaky(acc[s][r] + bias);
  }
  __syncthreads();

  // GEMM2: h1 @ w2 -> h2 (reuse msg buffer), leaky relu
  _Float16* h2 = msg;
  for (int nt = 0; nt < 8; ++nt) {
    v8f acc[4];
#pragma unroll
    for (int s = 0; s < 4; ++s) acc[s] = v8f{0.f,0.f,0.f,0.f,0.f,0.f,0.f,0.f};
#pragma unroll
    for (int kc = 0; kc < 4; ++kc) {
      v16h b = load_b_frag(wp, FR_W2 + kc * 8 + nt, lane);
#pragma unroll
      for (int s = 0; s < 4; ++s) {
        v16h a = load_a_frag(h1 + s * 16 * 128, 128, kc * 32, lane);
        acc[s] = wmma_f16(a, b, acc[s]);
      }
    }
    float bias = b2[nt * 16 + ln];
#pragma unroll
    for (int s = 0; s < 4; ++s)
#pragma unroll
      for (int r = 0; r < 8; ++r)
        h2[(s * 16 + r + hl * 8) * 128 + nt * 16 + ln] =
            (_Float16)leaky(acc[s][r] + bias);
  }
  __syncthreads();

  // GEMM3: h2 @ w3, scale by attention, scatter into dh
  for (int nt = 0; nt < 8; ++nt) {
    v8f acc[4];
#pragma unroll
    for (int s = 0; s < 4; ++s) acc[s] = v8f{0.f,0.f,0.f,0.f,0.f,0.f,0.f,0.f};
#pragma unroll
    for (int kc = 0; kc < 4; ++kc) {
      v16h b = load_b_frag(wp, FR_W3 + kc * 8 + nt, lane);
#pragma unroll
      for (int s = 0; s < 4; ++s) {
        v16h a = load_a_frag(h2 + s * 16 * 128, 128, kc * 32, lane);
        acc[s] = wmma_f16(a, b, acc[s]);
      }
    }
    float bias = b3[nt * 16 + ln];
#pragma unroll
    for (int s = 0; s < 4; ++s)
#pragma unroll
      for (int r = 0; r < 8; ++r) {
        int m = s * 16 + r + hl * 8;
        float val = (acc[s][r] + bias) * s_scale[m];
        atomicAdd(dh + (size_t)s_dst[m] * HDIM + nt * 16 + ln, val);
      }
  }
}

// ---------------------------------------------------------------------------
// K3: one wave per 16-node tile. x = h_V + dh; LN1; FFN (wi,relu,wo); +res;
// LN2; write out.
// ---------------------------------------------------------------------------
__global__ void __launch_bounds__(64)
node_kernel(const float* __restrict__ h_V, const float* __restrict__ dh,
            const float* __restrict__ bi, const float* __restrict__ bo,
            const float* __restrict__ g1, const float* __restrict__ be1,
            const float* __restrict__ g2, const float* __restrict__ be2,
            const _Float16* __restrict__ wp, float* __restrict__ out) {
  const int lane = (int)(threadIdx.x & 31);
  const int wv   = (int)(threadIdx.x >> 5);
  const int tile = (int)(blockIdx.x * 2 + wv);   // 1250 tiles exactly
  const int hl = lane >> 4, ln = lane & 15;

  __shared__ float    s_x[2][16 * 128];
  __shared__ _Float16 s_h[2][16 * 128];
  __shared__ _Float16 s_y[2][16 * 512];
  float*    x   = s_x[wv];
  _Float16* h1h = s_h[wv];
  _Float16* y   = s_y[wv];

  // x = h_V + dh
  for (int m = 0; m < 16; ++m) {
    int node = tile * 16 + m;
    float4 a = *reinterpret_cast<const float4*>(h_V + (size_t)node * HDIM + lane * 4);
    float4 b = *reinterpret_cast<const float4*>(dh + (size_t)node * HDIM + lane * 4);
    float* row = x + m * 128 + lane * 4;
    row[0] = a.x + b.x; row[1] = a.y + b.y; row[2] = a.z + b.z; row[3] = a.w + b.w;
  }
  __syncthreads();

  // LN1 -> x (f32) and h1h (f16)
  if (lane < 16) {
    int m = lane;
    float mu = 0.f;
    for (int c = 0; c < 128; ++c) mu += x[m * 128 + c];
    mu *= (1.f / 128.f);
    float var = 0.f;
    for (int c = 0; c < 128; ++c) { float d = x[m * 128 + c] - mu; var += d * d; }
    var *= (1.f / 128.f);
    float rstd = rsqrtf(var + 1e-6f);
    for (int c = 0; c < 128; ++c) {
      float h = g1[c] * (x[m * 128 + c] - mu) * rstd + be1[c];
      x[m * 128 + c]   = h;
      h1h[m * 128 + c] = (_Float16)h;
    }
  }
  __syncthreads();

  // FFN in: [16x128] @ wi[128x512], relu -> y f16
  for (int nt = 0; nt < 32; ++nt) {
    v8f acc = {0.f, 0.f, 0.f, 0.f, 0.f, 0.f, 0.f, 0.f};
#pragma unroll
    for (int kc = 0; kc < 4; ++kc) {
      v16h a = load_a_frag(h1h, 128, kc * 32, lane);
      v16h b = load_b_frag(wp, FR_WI + kc * 32 + nt, lane);
      acc = wmma_f16(a, b, acc);
    }
    float bias = bi[nt * 16 + ln];
#pragma unroll
    for (int r = 0; r < 8; ++r) {
      float v = acc[r] + bias;
      y[(r + hl * 8) * 512 + nt * 16 + ln] = (_Float16)(v > 0.f ? v : 0.f);
    }
  }
  __syncthreads();

  // FFN out: y @ wo[512x128], + bias + residual -> x (f32)
  for (int nt = 0; nt < 8; ++nt) {
    v8f acc = {0.f, 0.f, 0.f, 0.f, 0.f, 0.f, 0.f, 0.f};
#pragma unroll
    for (int kc = 0; kc < 16; ++kc) {
      v16h a = load_a_frag(y, 512, kc * 32, lane);
      v16h b = load_b_frag(wp, FR_WO + kc * 8 + nt, lane);
      acc = wmma_f16(a, b, acc);
    }
    float bias = bo[nt * 16 + ln];
#pragma unroll
    for (int r = 0; r < 8; ++r) {
      int M = r + hl * 8;
      int c = nt * 16 + ln;
      x[M * 128 + c] = x[M * 128 + c] + acc[r] + bias;
    }
  }
  __syncthreads();

  // LN2 -> out
  if (lane < 16) {
    int m = lane;
    int node = tile * 16 + m;
    float mu = 0.f;
    for (int c = 0; c < 128; ++c) mu += x[m * 128 + c];
    mu *= (1.f / 128.f);
    float var = 0.f;
    for (int c = 0; c < 128; ++c) { float d = x[m * 128 + c] - mu; var += d * d; }
    var *= (1.f / 128.f);
    float rstd = rsqrtf(var + 1e-6f);
    for (int c = 0; c < 128; ++c)
      out[(size_t)node * HDIM + c] =
          g2[c] * (x[m * 128 + c] - mu) * rstd + be2[c];
  }
}

// ---------------------------------------------------------------------------
extern "C" void kernel_launch(void* const* d_in, const int* in_sizes, int n_in,
                              void* d_out, int out_size, void* d_ws,
                              size_t ws_size, hipStream_t stream) {
  const float* h_V = (const float*)d_in[0];
  const float* h_E = (const float*)d_in[1];
  const float* w1  = (const float*)d_in[2];
  const float* b1  = (const float*)d_in[3];
  const float* w2  = (const float*)d_in[4];
  const float* b2  = (const float*)d_in[5];
  const float* w3  = (const float*)d_in[6];
  const float* b3  = (const float*)d_in[7];
  const float* Av  = (const float*)d_in[8];
  const float* wi  = (const float*)d_in[9];
  const float* bi  = (const float*)d_in[10];
  const float* wo  = (const float*)d_in[11];
  const float* bo  = (const float*)d_in[12];
  const float* g1  = (const float*)d_in[13];
  const float* be1 = (const float*)d_in[14];
  const float* g2  = (const float*)d_in[15];
  const float* be2 = (const float*)d_in[16];
  const int*   dst = (const int*)d_in[17];   // edge_idx[0] = first E entries

  char* ws = (char*)d_ws;
  _Float16* wp   = (_Float16*)(ws);
  float* denom   = (float*)(ws + DENOM_OFF);
  float* att     = (float*)(ws + ATT_OFF);
  float* dh      = (float*)(ws + DH_OFF);

  (void)hipMemsetAsync(denom, 0, NNODES * sizeof(float), stream);
  (void)hipMemsetAsync(dh, 0, (size_t)NNODES * HDIM * sizeof(float), stream);

  pack_weights<<<96, 128, 0, stream>>>(w1, w2, w3, wi, wo, wp);
  att_kernel<<<NEDGES / 8, 256, 0, stream>>>(h_V, h_E, Av, dst, att, denom);
  edge_kernel<<<NEDGES / 64, 32, 0, stream>>>(h_V, h_E, dst, b1, b2, b3,
                                              wp, att, denom, dh);
  node_kernel<<<(NNODES / 16) / 2, 64, 0, stream>>>(h_V, dh, bi, bo, g1, be1,
                                                    g2, be2, wp, (float*)d_out);
}
